// RoIHead_47502338294356
// MI455X (gfx1250) — matrix operand order repaired and checked by reading, not compiled, not run on hardware
//
#include <hip/hip_runtime.h>
#include <hip/hip_bf16.h>

typedef __attribute__((ext_vector_type(16))) _Float16 v16h;
typedef __attribute__((ext_vector_type(8)))  _Float16 v8h;
typedef __attribute__((ext_vector_type(8)))  float    v8f;

union V16 { v16h v; v8h h2[2]; _Float16 e[16]; };

#define NPTS   8192
#define RROI   128
#define GRD    6
#define NCELL  216           // G^3
#define MGRID  27648         // R * G^3
#define NSAMP  16
#define CIN    32
#define CMID   64
#define FCW    256
#define KPRE   13824         // G^3 * CMID
#define RAD2   0.16f
#define BNINV  0.9999950000374996f  // 1/sqrt(1+1e-5)

__device__ __forceinline__ v8f wmma32(v16h a, v16h b, v8f c) {
    return __builtin_amdgcn_wmma_f32_16x16x32_f16(
        false, a, false, b, (short)0, c, false, false);
}

// ---------------- weight conversion ----------------
__global__ void k_f32_to_f16(const float* __restrict__ src,
                             _Float16* __restrict__ dst, int n) {
    int i = blockIdx.x * blockDim.x + threadIdx.x;
    if (i < n) dst[i] = (_Float16)src[i];
}

// mlp_w0 [64][35] -> [64][64] zero-padded f16
__global__ void k_pad_w0(const float* __restrict__ src,
                         _Float16* __restrict__ dst) {
    int i = blockIdx.x * blockDim.x + threadIdx.x;
    if (i < 64 * 64) {
        int n = i >> 6, k = i & 63;
        dst[i] = (_Float16)(k < (3 + CIN) ? src[n * 35 + k] : 0.0f);
    }
}

// ---------------- grid points ----------------
__global__ void k_grid_points(const float* __restrict__ rois,
                              float* __restrict__ newxyz) {
    int m = blockIdx.x * blockDim.x + threadIdx.x;
    if (m >= MGRID) return;
    int r = m / NCELL, f = m % NCELL;
    int ix = f / 36, iy = (f / 6) % 6, iz = f % 6;
    const float* ro = rois + r * 7;
    float sx = ro[3], sy = ro[4], sz = ro[5], yaw = ro[6];
    float lx = (ix + 0.5f) * (1.0f / GRD) * sx - 0.5f * sx;
    float ly = (iy + 0.5f) * (1.0f / GRD) * sy - 0.5f * sy;
    float lz = (iz + 0.5f) * (1.0f / GRD) * sz - 0.5f * sz;
    float c = __cosf(yaw), s = __sinf(yaw);
    newxyz[m * 3 + 0] = lx * c - ly * s + ro[0];
    newxyz[m * 3 + 1] = lx * s + ly * c + ro[1];
    newxyz[m * 3 + 2] = lz + ro[2];
}

// ---------------- ball query (first-16 by index) ----------------
#define BQ_TILE 1024
__global__ __launch_bounds__(256)
void k_ball_query(const float* __restrict__ xyz,
                  const float* __restrict__ newxyz,
                  int* __restrict__ idx,
                  int* __restrict__ emptyf) {
    __shared__ float sx[BQ_TILE], sy[BQ_TILE], sz[BQ_TILE];
    int m = blockIdx.x * 256 + threadIdx.x;     // MGRID == 108*256 exactly
    float cx = newxyz[m * 3], cy = newxyz[m * 3 + 1], cz = newxyz[m * 3 + 2];
    int count = 0, first = 0;
    for (int t = 0; t < NPTS / BQ_TILE; ++t) {
        for (int i = threadIdx.x; i < BQ_TILE; i += 256) {
            int p = t * BQ_TILE + i;
            sx[i] = xyz[p * 3]; sy[i] = xyz[p * 3 + 1]; sz[i] = xyz[p * 3 + 2];
        }
        __syncthreads();
        for (int j = 0; j < BQ_TILE; ++j) {
            float dx = sx[j] - cx, dy = sy[j] - cy, dz = sz[j] - cz;
            float d2 = dx * dx + dy * dy + dz * dz;
            if (d2 < RAD2 && count < NSAMP) {
                int pi = t * BQ_TILE + j;
                if (count == 0) first = pi;
                idx[m * NSAMP + count] = pi;
                ++count;
            }
        }
        __syncthreads();
    }
    for (int s = count; s < NSAMP; ++s) idx[m * NSAMP + s] = first;
    emptyf[m] = (count == 0) ? 1 : 0;
}

// ---------------- group + MLP(35->64->64) + maxpool, 8 grid points per wave --
__global__ __launch_bounds__(256)
void k_mlp_pool(const float* __restrict__ xyz,
                const float* __restrict__ feat,
                const float* __restrict__ newxyz,
                const int* __restrict__ idx,
                const int* __restrict__ emptyf,
                const _Float16* __restrict__ w0,   // [64][64] padded
                const _Float16* __restrict__ w1,   // [64][64]
                const float* __restrict__ g0, const float* __restrict__ b0,
                const float* __restrict__ g1, const float* __restrict__ b1,
                _Float16* __restrict__ poolT) {    // [128][13824] = flat
    __shared__ __align__(32) _Float16 sW0[64 * 64];
    __shared__ __align__(32) _Float16 sW1[64 * 64];
    __shared__ __align__(32) _Float16 sH[8][16 * 64];

    int tid = threadIdx.x;
    for (int i = tid; i < 64 * 64; i += 256) { sW0[i] = w0[i]; sW1[i] = w1[i]; }
    __syncthreads();

    int wave = tid >> 5, lane = tid & 31;
    int hi = lane >> 4, ln = lane & 15;
    _Float16* hbuf = sH[wave];

    // B fragments: lane holds column n = 16t+ln, K at 32c+8hi+(e<8 ? e : e+8)
    V16 B1[4][2], B2[4][2];
    float G0[4], C0[4], G1[4], C1[4];
#pragma unroll
    for (int t = 0; t < 4; ++t) {
        int n = t * 16 + ln;
#pragma unroll
        for (int c = 0; c < 2; ++c) {
            int kb = c * 32 + hi * 8;
            const v8h* p0 = (const v8h*)(sW0 + n * 64 + kb);
            B1[t][c].h2[0] = p0[0]; B1[t][c].h2[1] = p0[2];
            const v8h* p1 = (const v8h*)(sW1 + n * 64 + kb);
            B2[t][c].h2[0] = p1[0]; B2[t][c].h2[1] = p1[2];
        }
        G0[t] = g0[n]; C0[t] = b0[n]; G1[t] = g1[n]; C1[t] = b1[n];
    }

    int wg = blockIdx.x * 8 + wave;
    for (int m = wg * 8; m < wg * 8 + 8; ++m) {
        int sIdx = idx[m * NSAMP + ln];          // row r = ln
        int isE  = emptyf[m];
        float cx = newxyz[m * 3], cy = newxyz[m * 3 + 1], cz = newxyz[m * 3 + 2];
        float px = xyz[sIdx * 3] - cx;
        float py = xyz[sIdx * 3 + 1] - cy;
        float pz = xyz[sIdx * 3 + 2] - cz;
        const float* frow = feat + sIdx * CIN;

        // A fragments: element e -> k = 32c + 8hi + (e<8 ? e : e+8)
        V16 A[2];
#pragma unroll
        for (int c = 0; c < 2; ++c) {
#pragma unroll
            for (int e = 0; e < 16; ++e) {
                int kk = c * 32 + hi * 8 + (e < 8 ? e : e + 8);
                float v;
                if (kk == 0) v = px;
                else if (kk == 1) v = py;
                else if (kk == 2) v = pz;
                else if (kk < 3 + CIN) v = frow[kk - 3];
                else v = 0.0f;
                if (isE) v = 0.0f;
                A[c].e[e] = (_Float16)v;
            }
        }

        // layer 1: interleave 4 independent tiles so WMMA->VALU hazards are
        // hidden by other tiles' WMMAs instead of v_nops
        v8f acc1[4];
#pragma unroll
        for (int t = 0; t < 4; ++t) { v8f z = {}; acc1[t] = wmma32(A[0].v, B1[t][0].v, z); }
#pragma unroll
        for (int t = 0; t < 4; ++t) { acc1[t] = wmma32(A[1].v, B1[t][1].v, acc1[t]); }
#pragma unroll
        for (int t = 0; t < 4; ++t) {
#pragma unroll
            for (int v = 0; v < 8; ++v) {
                float h = fmaxf(G0[t] * acc1[t][v] * BNINV + C0[t], 0.0f);
                hbuf[(v + hi * 8) * 64 + t * 16 + ln] = (_Float16)h;
            }
        }
        asm volatile("s_wait_dscnt 0" ::: "memory");

        // re-swizzle h into A layout for layer 2
        V16 A2[2];
#pragma unroll
        for (int c = 0; c < 2; ++c) {
            int base = c * 32 + hi * 8;
            A2[c].h2[0] = *(const v8h*)(hbuf + ln * 64 + base);
            A2[c].h2[1] = *(const v8h*)(hbuf + ln * 64 + base + 16);
        }

        // layer 2, same interleaving
        v8f acc2[4];
#pragma unroll
        for (int t = 0; t < 4; ++t) { v8f z = {}; acc2[t] = wmma32(A2[0].v, B2[t][0].v, z); }
#pragma unroll
        for (int t = 0; t < 4; ++t) { acc2[t] = wmma32(A2[1].v, B2[t][1].v, acc2[t]); }

        int roi = m / NCELL, cell = m % NCELL;
#pragma unroll
        for (int t = 0; t < 4; ++t) {
            float mv = 0.0f;                      // relu output >= 0
#pragma unroll
            for (int v = 0; v < 8; ++v) {
                float h = fmaxf(G1[t] * acc2[t][v] * BNINV + C1[t], 0.0f);
                mv = fmaxf(mv, h);
            }
            float o = __shfl_xor(mv, 16, 32);     // combine M halves
            mv = fmaxf(mv, o);
            if (hi == 0)
                poolT[roi * KPRE + (t * 16 + ln) * NCELL + cell] = (_Float16)mv;
        }
        asm volatile("s_wait_dscnt 0" ::: "memory");
    }
}

// ---------------- generic WMMA GEMM: out = act(A @ W^T), M=128 ------------
__global__ __launch_bounds__(256)
void k_gemm(const _Float16* __restrict__ A,   // [128][K]
            const _Float16* __restrict__ W,   // [N][K]
            _Float16* __restrict__ out,       // [128][N]
            int K, int N, int doRelu) {
    int wave = threadIdx.x >> 5, lane = threadIdx.x & 31;
    int hi = lane >> 4, ln = lane & 15;
    int ntiles = N >> 4;
    int tile = blockIdx.x * 8 + wave;
    int mt = tile / ntiles, nt = tile % ntiles;
    if (mt >= 8) return;
    const _Float16* arow = A + (size_t)(mt * 16 + ln) * K;
    const _Float16* wrow = W + (size_t)(nt * 16 + ln) * K;
    v8f acc = {};
    for (int k0 = 0; k0 < K; k0 += 32) {
        V16 a, b;
        a.h2[0] = *(const v8h*)(arow + k0 + hi * 8);
        a.h2[1] = *(const v8h*)(arow + k0 + hi * 8 + 16);
        b.h2[0] = *(const v8h*)(wrow + k0 + hi * 8);
        b.h2[1] = *(const v8h*)(wrow + k0 + hi * 8 + 16);
        acc = wmma32(a.v, b.v, acc);
    }
    int n = nt * 16 + ln;
#pragma unroll
    for (int v = 0; v < 8; ++v) {
        float x = acc[v];
        if (doRelu) x = fmaxf(x, 0.0f);
        out[(size_t)(mt * 16 + v + hi * 8) * N + n] = (_Float16)x;
    }
}

// ---------------- final 1/1/7 projections + bias ----------------
__global__ void k_heads_final(const _Float16* __restrict__ hc,
                              const _Float16* __restrict__ hio,
                              const _Float16* __restrict__ hr,
                              const float* __restrict__ cw, const float* __restrict__ cb,
                              const float* __restrict__ iw, const float* __restrict__ ib,
                              const float* __restrict__ rw, const float* __restrict__ rb,
                              float* __restrict__ out) {
    int r = threadIdx.x;   // 0..127
    int o = blockIdx.x;    // 0..8
    const _Float16* h; const float* w; float b;
    if (o == 0)      { h = hc  + r * FCW; w = cw;               b = cb[0]; }
    else if (o == 1) { h = hio + r * FCW; w = iw;               b = ib[0]; }
    else             { h = hr  + r * FCW; w = rw + (o - 2) * FCW; b = rb[o - 2]; }
    float s = b;
    for (int c = 0; c < FCW; ++c) s += (float)h[c] * w[c];
    if (o == 0)      out[r] = s;
    else if (o == 1) out[RROI + r] = s;
    else             out[2 * RROI + r * 7 + (o - 2)] = s;
}

extern "C" void kernel_launch(void* const* d_in, const int* in_sizes, int n_in,
                              void* d_out, int out_size, void* d_ws, size_t ws_size,
                              hipStream_t stream) {
    const float* xyz       = (const float*)d_in[0];
    const float* features  = (const float*)d_in[1];
    const float* rois      = (const float*)d_in[2];
    const float* mlp_w0    = (const float*)d_in[3];
    const float* mlp_g0    = (const float*)d_in[4];
    const float* mlp_b0    = (const float*)d_in[5];
    const float* mlp_w1    = (const float*)d_in[6];
    const float* mlp_g1    = (const float*)d_in[7];
    const float* mlp_b1    = (const float*)d_in[8];
    const float* shared_w0 = (const float*)d_in[9];
    const float* shared_w1 = (const float*)d_in[10];
    const float* cls_w0    = (const float*)d_in[11];
    const float* cls_w1    = (const float*)d_in[12];
    const float* cls_w2    = (const float*)d_in[13];
    const float* cls_b2    = (const float*)d_in[14];
    const float* iou_w0    = (const float*)d_in[15];
    const float* iou_w1    = (const float*)d_in[16];
    const float* iou_w2    = (const float*)d_in[17];
    const float* iou_b2    = (const float*)d_in[18];
    const float* reg_w0    = (const float*)d_in[19];
    const float* reg_w1    = (const float*)d_in[20];
    const float* reg_w2    = (const float*)d_in[21];
    const float* reg_b2    = (const float*)d_in[22];

    char* ws = (char*)d_ws;
    size_t off = 0;
    auto carve = [&](size_t bytes) { char* p = ws + off; off += (bytes + 255) & ~size_t(255); return p; };

    float*    newxyz  = (float*)   carve(MGRID * 3 * sizeof(float));
    int*      idx     = (int*)     carve(MGRID * NSAMP * sizeof(int));
    int*      emptyf  = (int*)     carve(MGRID * sizeof(int));
    _Float16* w0f     = (_Float16*)carve(64 * 64 * 2);
    _Float16* w1f     = (_Float16*)carve(64 * 64 * 2);
    _Float16* sw0f    = (_Float16*)carve((size_t)FCW * KPRE * 2);
    _Float16* sw1f    = (_Float16*)carve(FCW * FCW * 2);
    _Float16* cw0f    = (_Float16*)carve(FCW * FCW * 2);
    _Float16* cw1f    = (_Float16*)carve(FCW * FCW * 2);
    _Float16* iw0f    = (_Float16*)carve(FCW * FCW * 2);
    _Float16* iw1f    = (_Float16*)carve(FCW * FCW * 2);
    _Float16* rw0f    = (_Float16*)carve(FCW * FCW * 2);
    _Float16* rw1f    = (_Float16*)carve(FCW * FCW * 2);
    _Float16* poolT   = (_Float16*)carve((size_t)RROI * KPRE * 2);
    _Float16* x0      = (_Float16*)carve(RROI * FCW * 2);
    _Float16* x1      = (_Float16*)carve(RROI * FCW * 2);
    _Float16* hc1     = (_Float16*)carve(RROI * FCW * 2);
    _Float16* hc2     = (_Float16*)carve(RROI * FCW * 2);
    _Float16* hi1     = (_Float16*)carve(RROI * FCW * 2);
    _Float16* hi2     = (_Float16*)carve(RROI * FCW * 2);
    _Float16* hr1     = (_Float16*)carve(RROI * FCW * 2);
    _Float16* hr2     = (_Float16*)carve(RROI * FCW * 2);

    // weight conversion
    int nsw0 = FCW * KPRE;
    hipLaunchKernelGGL(k_f32_to_f16, dim3((nsw0 + 255) / 256), dim3(256), 0, stream, shared_w0, sw0f, nsw0);
    hipLaunchKernelGGL(k_f32_to_f16, dim3(256), dim3(256), 0, stream, shared_w1, sw1f, FCW * FCW);
    hipLaunchKernelGGL(k_f32_to_f16, dim3(256), dim3(256), 0, stream, cls_w0, cw0f, FCW * FCW);
    hipLaunchKernelGGL(k_f32_to_f16, dim3(256), dim3(256), 0, stream, cls_w1, cw1f, FCW * FCW);
    hipLaunchKernelGGL(k_f32_to_f16, dim3(256), dim3(256), 0, stream, iou_w0, iw0f, FCW * FCW);
    hipLaunchKernelGGL(k_f32_to_f16, dim3(256), dim3(256), 0, stream, iou_w1, iw1f, FCW * FCW);
    hipLaunchKernelGGL(k_f32_to_f16, dim3(256), dim3(256), 0, stream, reg_w0, rw0f, FCW * FCW);
    hipLaunchKernelGGL(k_f32_to_f16, dim3(256), dim3(256), 0, stream, reg_w1, rw1f, FCW * FCW);
    hipLaunchKernelGGL(k_pad_w0, dim3(16), dim3(256), 0, stream, mlp_w0, w0f);
    hipLaunchKernelGGL(k_f32_to_f16, dim3(16), dim3(256), 0, stream, mlp_w1, w1f, 64 * 64);

    // geometry
    hipLaunchKernelGGL(k_grid_points, dim3(MGRID / 256), dim3(256), 0, stream, rois, newxyz);
    hipLaunchKernelGGL(k_ball_query, dim3(MGRID / 256), dim3(256), 0, stream, xyz, newxyz, idx, emptyf);

    // per-point MLP + pool (WMMA)
    hipLaunchKernelGGL(k_mlp_pool, dim3(MGRID / (8 * 8)), dim3(256), 0, stream,
                       xyz, features, newxyz, idx, emptyf, w0f, w1f,
                       mlp_g0, mlp_b0, mlp_g1, mlp_b1, poolT);

    // FC stack (WMMA GEMMs): 128 tiles -> 16 blocks of 8 waves
    hipLaunchKernelGGL(k_gemm, dim3(16), dim3(256), 0, stream, poolT, sw0f, x0, KPRE, FCW, 1);
    hipLaunchKernelGGL(k_gemm, dim3(16), dim3(256), 0, stream, x0, sw1f, x1, FCW, FCW, 1);
    hipLaunchKernelGGL(k_gemm, dim3(16), dim3(256), 0, stream, x1, cw0f, hc1, FCW, FCW, 1);
    hipLaunchKernelGGL(k_gemm, dim3(16), dim3(256), 0, stream, hc1, cw1f, hc2, FCW, FCW, 1);
    hipLaunchKernelGGL(k_gemm, dim3(16), dim3(256), 0, stream, x1, iw0f, hi1, FCW, FCW, 1);
    hipLaunchKernelGGL(k_gemm, dim3(16), dim3(256), 0, stream, hi1, iw1f, hi2, FCW, FCW, 1);
    hipLaunchKernelGGL(k_gemm, dim3(16), dim3(256), 0, stream, x1, rw0f, hr1, FCW, FCW, 1);
    hipLaunchKernelGGL(k_gemm, dim3(16), dim3(256), 0, stream, hr1, rw1f, hr2, FCW, FCW, 1);

    // final projections
    hipLaunchKernelGGL(k_heads_final, dim3(9), dim3(128), 0, stream,
                       hc2, hi2, hr2, cls_w2, cls_b2, iou_w2, iou_b2,
                       reg_w2, reg_b2, (float*)d_out);
    (void)in_sizes; (void)n_in; (void)out_size; (void)ws_size;
}
